// NextVLAD_33913061769329
// MI455X (gfx1250) — matrix operand order repaired and checked by reading, not compiled, not run on hardware
//
#include <hip/hip_runtime.h>
#include <stdint.h>

typedef unsigned short bf16_t;
typedef __attribute__((ext_vector_type(16))) __bf16 v16bf;
typedef __attribute__((ext_vector_type(8)))  float  v8f;

// async-copy pointer types: int4 vectors in global (AS1) / LDS (AS3) address spaces
typedef int v4i_ __attribute__((vector_size(16)));
typedef __attribute__((address_space(1))) v4i_ as1v4i;
typedef __attribute__((address_space(3))) v4i_ as3v4i;

#if defined(__has_builtin)
#  if __has_builtin(__builtin_amdgcn_global_load_async_to_lds_b128) && \
      __has_builtin(__builtin_amdgcn_s_wait_asynccnt)
#    define CDNA5_ASYNC 1
#  endif
#endif
#ifndef CDNA5_ASYNC
#  define CDNA5_ASYNC 0
#endif

#if CDNA5_ASYNC
// generic -> AS1 : identical 64-bit value for global pointers
__device__ __forceinline__ as1v4i* to_gbl(const void* p) {
  return (as1v4i*)(unsigned long long)(uintptr_t)p;
}
// generic -> AS3 : hardware LDS address is addr[31:0] of the generic address
__device__ __forceinline__ as3v4i* to_lds(const void* p) {
  return (as3v4i*)(unsigned int)(uintptr_t)p;
}
#endif

// ---------- helpers ----------
__device__ __forceinline__ bf16_t f2bf(float f) {
  unsigned u = __float_as_uint(f);
  unsigned r = u + 0x7FFFu + ((u >> 16) & 1u);   // round-to-nearest-even
  return (bf16_t)(r >> 16);
}
__device__ __forceinline__ float wred_sum(float v) {
#pragma unroll
  for (int m = 16; m >= 1; m >>= 1) v += __shfl_xor(v, m, 32);
  return v;
}
__device__ __forceinline__ float wred_max(float v) {
#pragma unroll
  for (int m = 16; m >= 1; m >>= 1) v = fmaxf(v, __shfl_xor(v, m, 32));
  return v;
}

// ---------- elementwise f32 -> bf16 ----------
__global__ __launch_bounds__(256) void convert_bf16_kernel(const float* __restrict__ s,
                                                           bf16_t* __restrict__ d, int n) {
  int i = blockIdx.x * 256 + threadIdx.x;
  if (i < n) d[i] = f2bf(s[i]);
}

__global__ __launch_bounds__(256) void fill_zero_kernel(float* __restrict__ p, int n) {
  int i = blockIdx.x * 256 + threadIdx.x;
  if (i < n) p[i] = 0.0f;
}

// ---------- stage 1: permute + L2-normalize rows + pack bf16 ----------
// x: (64,1024,8,8) f32  ->  xn_bf: [bs=8][M=512][1024] bf16, row L2-normalized
__global__ __launch_bounds__(256) void normalize_pack_kernel(const float* __restrict__ x,
                                                             bf16_t* __restrict__ xn) {
  int row  = blockIdx.x * 8 + (threadIdx.x >> 5);   // 0..4095  (b*512+m)
  int lane = threadIdx.x & 31;
  int b = row >> 9, m = row & 511;
  int i8 = m >> 6, hw = m & 63;
  const float* base = x + ((size_t)(b * 8 + i8) * 1024) * 64 + hw;
  float vals[32];
  float ss = 0.f;
#pragma unroll
  for (int i = 0; i < 32; ++i) {
    int n = lane + i * 32;
    float v = base[(size_t)n * 64];
    vals[i] = v;
    ss += v * v;
  }
  ss = wred_sum(ss);
  float inv = 1.0f / fmaxf(sqrtf(ss), 1e-12f);
  bf16_t* dst = xn + (size_t)row * 1024;
#pragma unroll
  for (int i = 0; i < 32; ++i) dst[lane + i * 32] = f2bf(vals[i] * inv);
}

// ---------- WMMA bf16 GEMM, double-buffered LDS, async global->LDS staging ----------
// BMODE 0: B is weights [N x K] (C = A*B^T).
// BMODE 1: B is xr: logical [K_tot x N] where col(k_tot) = y_bf[b*512 + (k_tot&511)][(k_tot>>9)*256 + n]
// Batched via blockIdx.z with linear strides aZ/bZ/cZ (elements).
template <int BMODE, bool HASBIAS, bool WC, bool WBF>
__global__ __launch_bounds__(256) void gemm_bf16_kernel(
    const bf16_t* __restrict__ A, long long lda, long long aZ,
    const bf16_t* __restrict__ B, long long ldb, long long bZ,
    const float* __restrict__ bias,
    float* __restrict__ C, long long ldc, long long cZ,
    bf16_t* __restrict__ Cbf, int Kdim) {
  int z = blockIdx.z;
  A += (long long)z * aZ;
  B += (long long)z * bZ;
  if (WC) C += (long long)z * cZ;
  if (WBF) Cbf += (long long)z * cZ;

  // fragment-ordered LDS tiles, double buffered: per 16-wide tile, 32 lanes x 16 bf16
  __shared__ __align__(32) bf16_t lA[2][8][512];   // 16 KB
  __shared__ __align__(32) bf16_t lB[2][8][512];   // 16 KB

  int tid = threadIdx.x;
  int lane = tid & 31, wv = tid >> 5;
  int wave_m = wv >> 1, wave_n = wv & 1;        // 4x2 wave grid; each wave: 2x4 tiles
  long long mBase = (long long)blockIdx.x * 128;
  long long nBase = (long long)blockIdx.y * 128;

  v8f acc[2][4] = {};

  // async ops issued per wave per stage (A: 2, B(weight): 2)
  constexpr int AOPS = (BMODE == 0) ? 4 : 2;
  (void)AOPS;

  auto stage = [&](int buf, int k0) {
    // ---- A tile (128 x 32) ----
    {
      int r = tid >> 1;
      int kh16 = tid & 1;
      const bf16_t* src = A + (mBase + r) * lda + k0 + kh16 * 16;
      int tile = r >> 4, mm = r & 15;
#if CDNA5_ASYNC
      int off0 = tile * 1024 + (mm + kh16 * 16) * 32;
      char* lbase = (char*)&lA[buf][0][0];
      __builtin_amdgcn_global_load_async_to_lds_b128(
          to_gbl(src), to_lds(lbase + off0), 0, 0);
      __builtin_amdgcn_global_load_async_to_lds_b128(
          to_gbl(src + 8), to_lds(lbase + off0 + 16), 0, 0);
#else
      __builtin_prefetch(src + 32, 0, 0);
      bf16_t v[16] __attribute__((aligned(16)));
      *(uint4*)(v)     = *(const uint4*)(src);
      *(uint4*)(v + 8) = *(const uint4*)(src + 8);
#pragma unroll
      for (int i = 0; i < 16; ++i) {
        int k = kh16 * 16 + i;
        int hi = (k >> 4) & 1;
        int ln = mm + hi * 16;
        int j = k & 15;
        lA[buf][tile][ln * 16 + j] = v[i];
      }
#endif
    }
    // ---- B tile (32 x 128) ----
    if (BMODE == 0) {
      int n = tid >> 1;
      int kh16 = tid & 1;
      const bf16_t* src = B + (nBase + n) * ldb + k0 + kh16 * 16;
      int tile = n >> 4, nn = n & 15;
#if CDNA5_ASYNC
      int off0 = tile * 1024 + (nn + kh16 * 16) * 32;
      char* lbase = (char*)&lB[buf][0][0];
      __builtin_amdgcn_global_load_async_to_lds_b128(
          to_gbl(src), to_lds(lbase + off0), 0, 0);
      __builtin_amdgcn_global_load_async_to_lds_b128(
          to_gbl(src + 8), to_lds(lbase + off0 + 16), 0, 0);
#else
      __builtin_prefetch(src + 32, 0, 0);
      bf16_t v[16] __attribute__((aligned(16)));
      *(uint4*)(v)     = *(const uint4*)(src);
      *(uint4*)(v + 8) = *(const uint4*)(src + 8);
#pragma unroll
      for (int i = 0; i < 16; ++i) {
        int k = kh16 * 16 + i;
        int ln = nn + (k >> 4) * 16;
        lB[buf][tile][ln * 16 + (k & 15)] = v[i];
      }
#endif
    } else {
      // xr mode: per-element scatter (2B/lane) -> VGPR round trip
      int k = tid >> 3;
      int ns = (tid & 7) * 16;
      int g = k0 >> 9, m0 = k0 & 511;
      const bf16_t* src = B + (long long)(m0 + k) * ldb + g * 256 + nBase + ns;
      __builtin_prefetch(src + 32 * ldb, 0, 0);
      bf16_t v[16] __attribute__((aligned(16)));
      *(uint4*)(v)     = *(const uint4*)(src);
      *(uint4*)(v + 8) = *(const uint4*)(src + 8);
#pragma unroll
      for (int i = 0; i < 16; ++i) {
        int n = ns + i;
        int tile = n >> 4, nn = n & 15;
        int ln = nn + (k >> 4) * 16;
        lB[buf][tile][ln * 16 + (k & 15)] = v[i];
      }
    }
  };

  int nK = Kdim >> 5;
  stage(0, 0);
  for (int kt = 0; kt < nK; ++kt) {
    int buf = kt & 1;
    if (kt + 1 < nK) {
      stage(buf ^ 1, (kt + 1) * 32);
#if CDNA5_ASYNC
      __builtin_amdgcn_s_wait_asynccnt(AOPS);   // previous stage fully landed
#endif
    } else {
#if CDNA5_ASYNC
      __builtin_amdgcn_s_wait_asynccnt(0);
#endif
    }
    __syncthreads();
    // ---- compute: 2x4 wmma tiles per wave ----
    v16bf a0 = *(const v16bf*)(&lA[buf][wave_m * 2 + 0][lane * 16]);
    v16bf a1 = *(const v16bf*)(&lA[buf][wave_m * 2 + 1][lane * 16]);
#pragma unroll
    for (int j = 0; j < 4; ++j) {
      v16bf bf = *(const v16bf*)(&lB[buf][wave_n * 4 + j][lane * 16]);
      acc[0][j] = __builtin_amdgcn_wmma_f32_16x16x32_bf16(false, a0, false, bf,
                                                          (short)0, acc[0][j], false, false);
      acc[1][j] = __builtin_amdgcn_wmma_f32_16x16x32_bf16(false, a1, false, bf,
                                                          (short)0, acc[1][j], false, false);
    }
    __syncthreads();   // all reads of buf done before it is overwritten
  }

  // ---- epilogue ----
  int hi = lane >> 4, lcol = lane & 15;
#pragma unroll
  for (int i = 0; i < 2; ++i) {
#pragma unroll
    for (int j = 0; j < 4; ++j) {
      long long gn = nBase + (long long)(wave_n * 4 + j) * 16 + lcol;
      float bval = HASBIAS ? bias[gn] : 0.0f;
#pragma unroll
      for (int r = 0; r < 8; ++r) {
        long long gm = mBase + (long long)(wave_m * 2 + i) * 16 + r + hi * 8;
        float val = acc[i][j][r] + bval;
        long long idx = gm * ldc + gn;
        if (WC) C[idx] = val;
        if (WBF) Cbf[idx] = f2bf(val);
      }
    }
  }
}

// ---------- alpha_g = sigmoid(y @ W_g^T + b_g), one wave per row, y read as bf16 ----------
__global__ __launch_bounds__(256) void alpha_g_kernel(const bf16_t* __restrict__ ybf,
                                                      const float* __restrict__ Wg,
                                                      const float* __restrict__ bg,
                                                      float* __restrict__ alpha) {
  int row = blockIdx.x * 8 + (threadIdx.x >> 5);   // 0..4095
  int lane = threadIdx.x & 31;
  const unsigned* yr = (const unsigned*)(ybf + (size_t)row * 2048);
  float p[8] = {};
  for (int k2 = lane; k2 < 1024; k2 += 32) {
    unsigned pk = yr[k2];
    float x0 = __uint_as_float(pk << 16);
    float x1 = __uint_as_float(pk & 0xFFFF0000u);
    int k = k2 * 2;
#pragma unroll
    for (int g = 0; g < 8; ++g)
      p[g] += x0 * Wg[g * 2048 + k] + x1 * Wg[g * 2048 + k + 1];
  }
#pragma unroll
  for (int g = 0; g < 8; ++g) p[g] = wred_sum(p[g]);
  if (lane == 0) {
#pragma unroll
    for (int g = 0; g < 8; ++g)
      alpha[(size_t)row * 8 + g] = 1.0f / (1.0f + __expf(-(p[g] + bg[g])));
  }
}

// ---------- softmax over token dim M (axis=1), fused with alpha_g scale ----------
// Writes w^T bf16 at wT[b][k][g][m] = [((b*128+k)*8+g)*512+m]; accumulates sum_w[b*128+k].
__global__ __launch_bounds__(256) void softmax_weight_kernel(const float* __restrict__ logits,
                                                             const float* __restrict__ alpha,
                                                             bf16_t* __restrict__ wT,
                                                             float* __restrict__ sum_w) {
  int col  = blockIdx.x * 8 + (threadIdx.x >> 5);  // 0..8191  (b*1024+gk)
  int lane = threadIdx.x & 31;
  int b = col >> 10, gk = col & 1023;
  int g = gk >> 7, k = gk & 127;
  const float* base = logits + (size_t)b * 512 * 1024 + gk;
  float v[16];
  float mx = -3.4e38f;
#pragma unroll
  for (int i = 0; i < 16; ++i) {
    v[i] = base[(size_t)(lane + i * 32) * 1024];
    mx = fmaxf(mx, v[i]);
  }
  mx = wred_max(mx);
  float s = 0.f;
#pragma unroll
  for (int i = 0; i < 16; ++i) {
    v[i] = __expf(v[i] - mx);
    s += v[i];
  }
  s = wred_sum(s);
  float inv = 1.0f / s;
  bf16_t* wrow = wT + ((size_t)(b * 128 + k) * 8 + g) * 512;
  float cs = 0.f;
#pragma unroll
  for (int i = 0; i < 16; ++i) {
    int m = lane + i * 32;
    float w = v[i] * inv * alpha[((size_t)b * 512 + m) * 8 + g];
    wrow[m] = f2bf(w);
    cs += w;
  }
  cs = wred_sum(cs);
  if (lane == 0) atomicAdd(&sum_w[b * 128 + k], cs);
}

// ---------- vlad -= sum_w * centroids; L2-normalize each 256-vec (in place) ----------
__global__ __launch_bounds__(256) void vlad_finalize_kernel(float* __restrict__ vlad,
                                                            const float* __restrict__ sum_w,
                                                            const float* __restrict__ cent) {
  int row  = blockIdx.x * 8 + (threadIdx.x >> 5);  // 0..1023 (b*128+k)
  int lane = threadIdx.x & 31;
  int k = row & 127;
  float sw = sum_w[row];
  float* vr = vlad + (size_t)row * 256;
  float vals[8];
  float ss = 0.f;
#pragma unroll
  for (int i = 0; i < 8; ++i) {
    int d = lane + i * 32;
    float v = vr[d] - sw * cent[k * 256 + d];
    vals[i] = v;
    ss += v * v;
  }
  ss = wred_sum(ss);
  float inv = 1.0f / fmaxf(sqrtf(ss), 1e-12f);
#pragma unroll
  for (int i = 0; i < 8; ++i) vr[lane + i * 32] = vals[i] * inv;
}

// ---------- final L2 norm over 32768 per batch ----------
__global__ __launch_bounds__(256) void final_norm_kernel(const float* __restrict__ vlad,
                                                         float* __restrict__ out) {
  int b = blockIdx.x;
  const float* vr = vlad + (size_t)b * 32768;
  float ss = 0.f;
  for (int i = threadIdx.x; i < 32768; i += 256) {
    float v = vr[i];
    ss += v * v;
  }
  ss = wred_sum(ss);
  __shared__ float red[8];
  __shared__ float inv_s;
  int wv = threadIdx.x >> 5, lane = threadIdx.x & 31;
  if (lane == 0) red[wv] = ss;
  __syncthreads();
  if (threadIdx.x == 0) {
    float t = 0.f;
    for (int i = 0; i < 8; ++i) t += red[i];
    inv_s = 1.0f / fmaxf(sqrtf(t), 1e-12f);
  }
  __syncthreads();
  float inv = inv_s;
  float* o = out + (size_t)b * 32768;
  for (int i = threadIdx.x; i < 32768; i += 256) o[i] = vr[i] * inv;
}

// ---------- host launch ----------
extern "C" void kernel_launch(void* const* d_in, const int* in_sizes, int n_in,
                              void* d_out, int out_size, void* d_ws, size_t ws_size,
                              hipStream_t stream) {
  (void)in_sizes; (void)n_in; (void)out_size; (void)ws_size;
  const float* x     = (const float*)d_in[0];
  const float* W_inp = (const float*)d_in[1];
  const float* b_inp = (const float*)d_in[2];
  const float* W_g   = (const float*)d_in[3];
  const float* b_g   = (const float*)d_in[4];
  const float* W_gk  = (const float*)d_in[5];
  const float* b_gk  = (const float*)d_in[6];
  const float* cent  = (const float*)d_in[7];
  float* out = (float*)d_out;
  char* ws = (char*)d_ws;

  bf16_t* xn_bf   = (bf16_t*)(ws + 0);          //  8 MB  [4096][1024]
  bf16_t* winp_bf = (bf16_t*)(ws + 8388608);    //  4 MB  [2048][1024]
  bf16_t* wgk_bf  = (bf16_t*)(ws + 12582912);   //  4 MB  [1024][2048]
  bf16_t* y_bf    = (bf16_t*)(ws + 16777216);   // 16 MB  [4096][2048]
  float*  logits  = (float*)(ws + 33554432);    // 16 MB  [4096][1024]
  float*  alpha   = (float*)(ws + 50331648);    // 128 KB [4096][8]
  bf16_t* wT      = (bf16_t*)(ws + 50462720);   //  8 MB  [8][128][8][512]
  float*  sum_w   = (float*)(ws + 58851328);    //  4 KB  [1024]
  float*  vlad    = (float*)(ws + 58855424);    //  1 MB  [8][128][256]

  // weight converts
  convert_bf16_kernel<<<8192, 256, 0, stream>>>(W_inp, winp_bf, 2097152);
  convert_bf16_kernel<<<8192, 256, 0, stream>>>(W_gk, wgk_bf, 2097152);
  // permute + l2norm + pack
  normalize_pack_kernel<<<512, 256, 0, stream>>>(x, xn_bf);
  // GEMM1: y_bf = bf16(xn @ W_inp^T + b_inp)
  gemm_bf16_kernel<0, true, false, true><<<dim3(32, 16, 1), 256, 0, stream>>>(
      xn_bf, 1024, 0, winp_bf, 1024, 0, b_inp, nullptr, 2048, 0, y_bf, 1024);
  // alpha_g (from bf16 y)
  alpha_g_kernel<<<512, 256, 0, stream>>>(y_bf, W_g, b_g, alpha);
  // GEMM2: logits = y @ W_gk^T + b_gk
  gemm_bf16_kernel<0, true, true, false><<<dim3(32, 8, 1), 256, 0, stream>>>(
      y_bf, 2048, 0, wgk_bf, 2048, 0, b_gk, logits, 1024, 0, nullptr, 2048);
  // zero column-sum accumulator
  fill_zero_kernel<<<4, 256, 0, stream>>>(sum_w, 1024);
  // softmax over M + alpha_g scale -> w^T bf16 [b][k][g][m], column sums
  softmax_weight_kernel<<<1024, 256, 0, stream>>>(logits, alpha, wT, sum_w);
  // GEMM3 (batched over b, g fused into K=4096): vlad[b] = w^T[b] @ xr[b]
  gemm_bf16_kernel<1, false, true, false><<<dim3(1, 2, 8), 256, 0, stream>>>(
      wT, 4096, 524288,          // A: [b][128][4096]
      y_bf, 2048, 1048576,       // B: y_bf[b*512+m][g*256+d], segmented by k0>>9
      nullptr, vlad, 256, 32768, // C: vlad[b][k][d]
      nullptr, 4096);
  // subtract sum_w * centroids, intra-row L2 norm
  vlad_finalize_kernel<<<128, 256, 0, stream>>>(vlad, sum_w, cent);
  // final whole-vector L2 norm -> out
  final_norm_kernel<<<8, 256, 0, stream>>>(vlad, out);
}
